// MultiHeadAttentionLayer_45372034515444
// MI455X (gfx1250) — compile-verified, matrix operand order checked by reading, not run on hardware
//
#include <hip/hip_runtime.h>
#include <hip/hip_bf16.h>

// Problem constants
#define Bb 2
#define Ss 2048
#define Dd 1024
#define Hh 16
#define DK 64
#define Mrows (Bb * Ss)   // 4096

typedef __attribute__((ext_vector_type(16))) __bf16 v16bf;
typedef __attribute__((ext_vector_type(8)))  __bf16 v8bf;
typedef __attribute__((ext_vector_type(8)))  float  v8f;

__device__ __forceinline__ v8f wmma_bf16(v16bf a, v16bf b, v8f c) {
    return __builtin_amdgcn_wmma_f32_16x16x32_bf16(
        false, a, false, b, (short)0, c, false, false);
}

// 16-bit fragment = two contiguous 8-element runs at src[hi*8] and src[16+hi*8]
// (ISA 7.12.2 16-bit A 16x32 swizzle). Two 16-byte vector loads -> 2x b128.
__device__ __forceinline__ v16bf frag_ld(const __bf16* src, int hi) {
    v8bf lo = *(const v8bf*)(src + hi * 8);
    v8bf hb = *(const v8bf*)(src + 16 + hi * 8);
    v16bf a;
#pragma unroll
    for (int e = 0; e < 8; ++e) { a[e] = lo[e]; a[e + 8] = hb[e]; }
    return a;
}

// ---------------------------------------------------------------------------
// Elementwise fp32 -> bf16 (n multiple of 4).
// ---------------------------------------------------------------------------
__global__ __launch_bounds__(256)
void cvt_bf16_kernel(const float* __restrict__ in, __bf16* __restrict__ out, long n) {
    long i = ((long)blockIdx.x * 256 + threadIdx.x) * 4;
    if (i + 3 < n) {
        float4 f = *(const float4*)(in + i);
        __bf16 p[4] = {(__bf16)f.x, (__bf16)f.y, (__bf16)f.z, (__bf16)f.w};
        *(ulong1*)(out + i) = *(ulong1*)p;  // 8-byte store
    }
}

// ---------------------------------------------------------------------------
// Kernel 1: Y = X @ W^T + b (all bf16 in), out bf16 head-major.
//   transposeV == 0 : out[b][h][s][dk]   (Q, K)
//   transposeV == 1 : out[b][h][dk][s]   (V, for contiguous P.V B-fragments)
// One wave per 64x64 output tile. grid = (Dd/64, Mrows/64), block = 32.
// ---------------------------------------------------------------------------
__global__ __launch_bounds__(32)
void qkv_proj_kernel(const __bf16* __restrict__ x, const __bf16* __restrict__ W,
                     const float* __restrict__ bias, __bf16* __restrict__ out,
                     int transposeV) {
    const int lane  = threadIdx.x;
    const int hi    = lane >> 4;
    const int colid = lane & 15;
    const int nbase = blockIdx.x * 64;
    const int mbase = blockIdx.y * 64;

    const __bf16* arow[4];
    const __bf16* brow[4];
#pragma unroll
    for (int t = 0; t < 4; ++t) {
        arow[t] = x + (long)(mbase + t * 16 + colid) * Dd;
        brow[t] = W + (long)(nbase + t * 16 + colid) * Dd;
    }

    v8f acc[4][4];
#pragma unroll
    for (int i = 0; i < 4; ++i)
#pragma unroll
        for (int j = 0; j < 4; ++j) acc[i][j] = (v8f)0.0f;

    for (int k0 = 0; k0 < Dd; k0 += 32) {
        if (k0 + 32 < Dd) {
            __builtin_prefetch(arow[0] + k0 + 32, 0, 1);
            __builtin_prefetch(brow[0] + k0 + 32, 0, 1);
        }
        v16bf af[4];
#pragma unroll
        for (int mt = 0; mt < 4; ++mt) af[mt] = frag_ld(arow[mt] + k0, hi);
#pragma unroll
        for (int nt = 0; nt < 4; ++nt) {
            v16bf bf = frag_ld(brow[nt] + k0, hi);
#pragma unroll
            for (int mt = 0; mt < 4; ++mt)
                acc[mt][nt] = wmma_bf16(af[mt], bf, acc[mt][nt]);
        }
    }

#pragma unroll
    for (int mt = 0; mt < 4; ++mt) {
#pragma unroll
        for (int nt = 0; nt < 4; ++nt) {
#pragma unroll
            for (int r = 0; r < 8; ++r) {
                int row = mbase + mt * 16 + r + hi * 8;   // b*S + s
                int col = nbase + nt * 16 + colid;        // h*DK + dk
                int b = row >> 11, s = row & (Ss - 1);
                int h = col >> 6,  dk = col & (DK - 1);
                __bf16 v = (__bf16)(acc[mt][nt][r] + bias[col]);
                long idx = transposeV
                    ? (((long)b * Hh + h) * DK + dk) * Ss + s
                    : (((long)b * Hh + h) * Ss + s) * DK + dk;
                out[idx] = v;
            }
        }
    }
}

// ---------------------------------------------------------------------------
// Kernel 2: causal flash attention, 64-query tile per wave (4 M-subtiles).
// Q,K: [B,H,S,DK] bf16; Vt: [B,H,DK,S] bf16. grid = (Ss/64, Bb*Hh), block=32.
// K fragments reused across 4 query subtiles; V fragments reused across 4 P
// fragments -> 32 WMMA per 32-key step against 16 b128 loads.
// ---------------------------------------------------------------------------
__global__ __launch_bounds__(32)
void flash_attn_kernel(const __bf16* __restrict__ Q, const __bf16* __restrict__ K,
                       const __bf16* __restrict__ Vt, __bf16* __restrict__ xo) {
    __shared__ __bf16 Pl[4][16 * 32];

    const int lane  = threadIdx.x;
    const int hi    = lane >> 4;
    const int colid = lane & 15;
    const int q0    = blockIdx.x * 64;
    const int bh    = blockIdx.y;
    const int bidx  = bh >> 4;
    const int hidx  = bh & 15;

    const __bf16* Qh = Q  + (long)bh * Ss * DK;
    const __bf16* Kh = K  + (long)bh * Ss * DK;
    const __bf16* Vh = Vt + (long)bh * DK * Ss;

    v16bf qa[4][2];
#pragma unroll
    for (int mt = 0; mt < 4; ++mt) {
        const __bf16* qrow = Qh + (long)(q0 + mt * 16 + colid) * DK;
        qa[mt][0] = frag_ld(qrow + 0, hi);
        qa[mt][1] = frag_ld(qrow + 32, hi);
    }

    v8f acc[4][4];
#pragma unroll
    for (int i = 0; i < 4; ++i)
#pragma unroll
        for (int j = 0; j < 4; ++j) acc[i][j] = (v8f)0.0f;
    float m_i[4][8], l_i[4][8];
#pragma unroll
    for (int mt = 0; mt < 4; ++mt)
#pragma unroll
        for (int r = 0; r < 8; ++r) { m_i[mt][r] = -3.0e38f; l_i[mt][r] = 0.0f; }

    const float scale = 0.125f;  // 1/sqrt(DK)
    const int jend = q0 + 64;    // causal

    for (int j0 = 0; j0 < jend; j0 += 32) {
        // ---- scores for all 4 query subtiles; K fragments loaded once ----
        v8f sc[4][2];
#pragma unroll
        for (int jt = 0; jt < 2; ++jt) {
            const __bf16* krow = Kh + (long)(j0 + jt * 16 + colid) * DK;
            v16bf kb0 = frag_ld(krow + 0, hi);
            v16bf kb1 = frag_ld(krow + 32, hi);
#pragma unroll
            for (int mt = 0; mt < 4; ++mt) {
                v8f c = (v8f)0.0f;
                c = wmma_bf16(qa[mt][0], kb0, c);
                c = wmma_bf16(qa[mt][1], kb1, c);
                sc[mt][jt] = c;
            }
        }

        // ---- scale + causal mask + online softmax (per subtile) ----
#pragma unroll
        for (int mt = 0; mt < 4; ++mt) {
            float corr[8];
#pragma unroll
            for (int r = 0; r < 8; ++r) {
                int q = q0 + mt * 16 + r + hi * 8;
                float v0 = (j0 + colid      > q) ? -1.0e9f : sc[mt][0][r] * scale;
                float v1 = (j0 + 16 + colid > q) ? -1.0e9f : sc[mt][1][r] * scale;
                float mx = fmaxf(v0, v1);
#pragma unroll
                for (int msk = 1; msk < 16; msk <<= 1)
                    mx = fmaxf(mx, __shfl_xor(mx, msk, 32));
                float mnew = fmaxf(m_i[mt][r], mx);
                float cr = __expf(m_i[mt][r] - mnew);
                m_i[mt][r] = mnew;
                float e0 = __expf(v0 - mnew);
                float e1 = __expf(v1 - mnew);
                sc[mt][0][r] = e0;
                sc[mt][1][r] = e1;
                float rs = e0 + e1;
#pragma unroll
                for (int msk = 1; msk < 16; msk <<= 1)
                    rs += __shfl_xor(rs, msk, 32);
                l_i[mt][r] = l_i[mt][r] * cr + rs;
                corr[r] = cr;
            }
#pragma unroll
            for (int dt = 0; dt < 4; ++dt)
#pragma unroll
                for (int r = 0; r < 8; ++r) acc[mt][dt][r] *= corr[r];
        }

        // ---- P: C fragments -> A fragments via LDS bounce ----
        __syncthreads();  // single-wave WG: cheap dscnt fence
#pragma unroll
        for (int mt = 0; mt < 4; ++mt)
#pragma unroll
            for (int jt = 0; jt < 2; ++jt)
#pragma unroll
                for (int r = 0; r < 8; ++r)
                    Pl[mt][(r + hi * 8) * 32 + colid + jt * 16] = (__bf16)sc[mt][jt][r];
        __syncthreads();
        v16bf pa[4];
#pragma unroll
        for (int mt = 0; mt < 4; ++mt) pa[mt] = frag_ld(&Pl[mt][colid * 32], hi);

        // ---- acc += P(16x32) . V(32x64); V fragments loaded once per dt ----
#pragma unroll
        for (int dt = 0; dt < 4; ++dt) {
            v16bf vb = frag_ld(Vh + (long)(dt * 16 + colid) * Ss + j0, hi);
#pragma unroll
            for (int mt = 0; mt < 4; ++mt)
                acc[mt][dt] = wmma_bf16(pa[mt], vb, acc[mt][dt]);
        }
    }

    // ---- normalize + store bf16 to [B*S, D] ----
#pragma unroll
    for (int mt = 0; mt < 4; ++mt) {
#pragma unroll
        for (int dt = 0; dt < 4; ++dt) {
#pragma unroll
            for (int r = 0; r < 8; ++r) {
                int s = q0 + mt * 16 + r + hi * 8;
                long row = (long)bidx * Ss + s;
                int col = hidx * DK + dt * 16 + colid;
                xo[row * Dd + col] = (__bf16)(acc[mt][dt][r] / l_i[mt][r]);
            }
        }
    }
}

// ---------------------------------------------------------------------------
// Kernel 3: Y = X @ Wo^T + bo, bf16 in, fp32 out [B,S,D].
// grid = (Dd/64, Mrows/64), block = 32.
// ---------------------------------------------------------------------------
__global__ __launch_bounds__(32)
void out_proj_kernel(const __bf16* __restrict__ x, const __bf16* __restrict__ W,
                     const float* __restrict__ bias, float* __restrict__ y) {
    const int lane  = threadIdx.x;
    const int hi    = lane >> 4;
    const int colid = lane & 15;
    const int nbase = blockIdx.x * 64;
    const int mbase = blockIdx.y * 64;

    const __bf16* arow[4];
    const __bf16* brow[4];
#pragma unroll
    for (int t = 0; t < 4; ++t) {
        arow[t] = x + (long)(mbase + t * 16 + colid) * Dd;
        brow[t] = W + (long)(nbase + t * 16 + colid) * Dd;
    }

    v8f acc[4][4];
#pragma unroll
    for (int i = 0; i < 4; ++i)
#pragma unroll
        for (int j = 0; j < 4; ++j) acc[i][j] = (v8f)0.0f;

    for (int k0 = 0; k0 < Dd; k0 += 32) {
        if (k0 + 32 < Dd) {
            __builtin_prefetch(arow[0] + k0 + 32, 0, 1);
            __builtin_prefetch(brow[0] + k0 + 32, 0, 1);
        }
        v16bf af[4];
#pragma unroll
        for (int mt = 0; mt < 4; ++mt) af[mt] = frag_ld(arow[mt] + k0, hi);
#pragma unroll
        for (int nt = 0; nt < 4; ++nt) {
            v16bf bf = frag_ld(brow[nt] + k0, hi);
#pragma unroll
            for (int mt = 0; mt < 4; ++mt)
                acc[mt][nt] = wmma_bf16(af[mt], bf, acc[mt][nt]);
        }
    }

#pragma unroll
    for (int mt = 0; mt < 4; ++mt) {
#pragma unroll
        for (int nt = 0; nt < 4; ++nt) {
#pragma unroll
            for (int r = 0; r < 8; ++r) {
                int row = mbase + mt * 16 + r + hi * 8;
                int col = nbase + nt * 16 + colid;
                y[(long)row * Dd + col] = acc[mt][nt][r] + bias[col];
            }
        }
    }
}

extern "C" void kernel_launch(void* const* d_in, const int* in_sizes, int n_in,
                              void* d_out, int out_size, void* d_ws, size_t ws_size,
                              hipStream_t stream) {
    const float* query = (const float*)d_in[0];
    const float* key_  = (const float*)d_in[1];
    const float* value = (const float*)d_in[2];
    // d_in[3] = mask (causality handled analytically)
    const float* Wq = (const float*)d_in[4];
    const float* bq = (const float*)d_in[5];
    const float* Wk = (const float*)d_in[6];
    const float* bk = (const float*)d_in[7];
    const float* Wv = (const float*)d_in[8];
    const float* bv = (const float*)d_in[9];
    const float* Wo = (const float*)d_in[10];
    const float* bo = (const float*)d_in[11];
    float* out = (float*)d_out;

    const long nAct = (long)Mrows * Dd;   // 4M elements
    const long nWgt = (long)Dd * Dd;      // 1M elements

    // Workspace layout (bf16 elements)
    __bf16* Xq  = (__bf16*)d_ws;          // bf16 inputs
    __bf16* Xk  = Xq  + nAct;
    __bf16* Xv  = Xk  + nAct;
    __bf16* Wqb = Xv  + nAct;             // bf16 weights
    __bf16* Wkb = Wqb + nWgt;
    __bf16* Wvb = Wkb + nWgt;
    __bf16* Wob = Wvb + nWgt;
    __bf16* Qh  = Wob + nWgt;             // head-major projections
    __bf16* Kh  = Qh  + nAct;
    __bf16* Vth = Kh  + nAct;             // V transposed per head
    __bf16* Xat = Vth + nAct;             // attention output [B*S, D]

    dim3 b256(256);
    cvt_bf16_kernel<<<dim3(nAct / 1024), b256, 0, stream>>>(query, Xq, nAct);
    cvt_bf16_kernel<<<dim3(nAct / 1024), b256, 0, stream>>>(key_,  Xk, nAct);
    cvt_bf16_kernel<<<dim3(nAct / 1024), b256, 0, stream>>>(value, Xv, nAct);
    cvt_bf16_kernel<<<dim3(nWgt / 1024), b256, 0, stream>>>(Wq, Wqb, nWgt);
    cvt_bf16_kernel<<<dim3(nWgt / 1024), b256, 0, stream>>>(Wk, Wkb, nWgt);
    cvt_bf16_kernel<<<dim3(nWgt / 1024), b256, 0, stream>>>(Wv, Wvb, nWgt);
    cvt_bf16_kernel<<<dim3(nWgt / 1024), b256, 0, stream>>>(Wo, Wob, nWgt);

    dim3 blk(32);
    dim3 gProj(Dd / 64, Mrows / 64);       // (16, 64)
    qkv_proj_kernel<<<gProj, blk, 0, stream>>>(Xq, Wqb, bq, Qh, 0);
    qkv_proj_kernel<<<gProj, blk, 0, stream>>>(Xk, Wkb, bk, Kh, 0);
    qkv_proj_kernel<<<gProj, blk, 0, stream>>>(Xv, Wvb, bv, Vth, 1);

    dim3 gAttn(Ss / 64, Bb * Hh);          // (32, 32)
    flash_attn_kernel<<<gAttn, blk, 0, stream>>>(Qh, Kh, Vth, Xat);

    out_proj_kernel<<<gProj, blk, 0, stream>>>(Xat, Wob, bo, out);
}
